// TraceableEncoderLayer_66709432041881
// MI455X (gfx1250) — matrix operand order repaired and checked by reading, not compiled, not run on hardware
//
#include <hip/hip_runtime.h>
#include <hip/hip_bf16.h>

typedef __attribute__((ext_vector_type(16))) _Float16 v16h;
typedef __attribute__((ext_vector_type(8)))  _Float16 v8h;
typedef __attribute__((ext_vector_type(8)))  float    v8f;

#define SHUF16(lo, hh) __builtin_shufflevector((lo), (hh), 0,1,2,3,4,5,6,7,8,9,10,11,12,13,14,15)

static constexpr int Bb  = 2;
static constexpr int Tt  = 2048;
static constexpr int Dd  = 1024;
static constexpr int Hh  = 16;
static constexpr int DH  = 64;
static constexpr int DFF = 4096;
static constexpr int Mrows = Bb * Tt;           // 4096

#define WMMA_F16(a, b, c) __builtin_amdgcn_wmma_f32_16x16x32_f16(false, (a), false, (b), (short)0, (c), false, false)

// ---- CDNA5 async global->LDS DMA (ASYNCcnt path) ----------------------------
__device__ __forceinline__ void async_ld_b128(unsigned int lds_off, const _Float16* g) {
  asm volatile("global_load_async_to_lds_b128 %0, %1, off"
               :: "v"(lds_off), "v"(g) : "memory");
}
__device__ __forceinline__ void wait_async0() {
  asm volatile("s_wait_asynccnt 0" ::: "memory");
}

// ---------------------------------------------------------------- f32 -> f16
__global__ __launch_bounds__(256) void f32_to_f16_kernel(const float* __restrict__ in,
                                                         _Float16* __restrict__ out, int n) {
  int i = blockIdx.x * 256 + threadIdx.x;
  if (i < n) out[i] = (_Float16)in[i];
}

// ---------------------------------------------------------------- LayerNorm
template <bool USE_MASK>
__global__ __launch_bounds__(256) void ln_kernel(const float* __restrict__ x,
                                                 const float* __restrict__ w,
                                                 const float* __restrict__ mask,
                                                 _Float16* __restrict__ out) {
  const int row = blockIdx.x;
  const float* xr = x + (size_t)row * Dd;
  __shared__ float red[256];
  const int tid = threadIdx.x;

  float s = 0.f;
  for (int i = tid; i < Dd; i += 256) s += xr[i];
  red[tid] = s; __syncthreads();
  for (int st = 128; st > 0; st >>= 1) { if (tid < st) red[tid] += red[tid + st]; __syncthreads(); }
  float mu = red[0] / Dd;
  __syncthreads();

  float v = 0.f;
  for (int i = tid; i < Dd; i += 256) { float d = xr[i] - mu; v += d * d; }
  red[tid] = v; __syncthreads();
  for (int st = 128; st > 0; st >>= 1) { if (tid < st) red[tid] += red[tid + st]; __syncthreads(); }
  float var = red[0] / Dd;

  float rstd = rsqrtf(var + 1e-5f);
  float mv = USE_MASK ? mask[row] : 1.f;
  for (int i = tid; i < Dd; i += 256)
    out[(size_t)row * Dd + i] = (_Float16)((xr[i] - mu) * rstd * w[i] * mv);
}

// ---------------------------------------------------------------- WMMA GEMM
// C[M,N] = A[M,K] @ W[N,K]^T, f16 in, f32 accum.
// Block tile 128x128: 8 waves of 32x64 (acc[2][4]), BK=32.
// Tiles staged to LDS by GLOBAL_LOAD_ASYNC_TO_LDS_B128, double-buffered so the
// DMA of tile k+1 overlaps the 64 block-wide WMMAs on tile k.
// EPI: 1 = f16 store; 2 = gelu*mask -> f16; 3 = res + acc -> f32; 4 = res + acc*mask -> f32
__device__ __forceinline__ float gelu_tanh(float v) {
  return 0.5f * v * (1.0f + tanhf(0.7978845608028654f * (v + 0.044715f * v * v * v)));
}

template <int EPI>
__global__ __launch_bounds__(256) void gemm_wmma(const _Float16* __restrict__ A,
                                                 const _Float16* __restrict__ W,
                                                 void* __restrict__ outv,
                                                 const float* __restrict__ res,
                                                 const float* __restrict__ mask,
                                                 int M, int N, int K) {
  __shared__ _Float16 As[2][128 * 32];
  __shared__ _Float16 Bs[2][128 * 32];
  const int tid = threadIdx.x, lane = tid & 31, wave = tid >> 5;
  const int wm = (wave >> 1) * 32;      // 4 waves along M
  const int wn = (wave & 1) * 64;       // 2 waves along N
  const int lr = lane & 15, hf = lane >> 4;
  const int r0 = blockIdx.y * 128, n0 = blockIdx.x * 128;

  v8f acc[2][4];
  for (int i = 0; i < 2; ++i)
    for (int j = 0; j < 4; ++j) { v8f z = {}; acc[i][j] = z; }

  // each thread DMAs 4x16B chunks per tile (A:2, B:2); chunk = 8 halves
  const int sr = tid >> 2;              // 0..63
  const int sc = (tid & 3) * 8;         // 0,8,16,24
  auto issue_tile = [&](int k0, int buf) {
    async_ld_b128((unsigned int)(uintptr_t)&As[buf][sr * 32 + sc],
                  A + (size_t)(r0 + sr) * K + k0 + sc);
    async_ld_b128((unsigned int)(uintptr_t)&As[buf][(sr + 64) * 32 + sc],
                  A + (size_t)(r0 + sr + 64) * K + k0 + sc);
    async_ld_b128((unsigned int)(uintptr_t)&Bs[buf][sr * 32 + sc],
                  W + (size_t)(n0 + sr) * K + k0 + sc);
    async_ld_b128((unsigned int)(uintptr_t)&Bs[buf][(sr + 64) * 32 + sc],
                  W + (size_t)(n0 + sr + 64) * K + k0 + sc);
  };

  issue_tile(0, 0);
  wait_async0();
  __syncthreads();

  for (int k0 = 0; k0 < K; k0 += 32) {
    const int cur = (k0 >> 5) & 1;
    if (k0 + 32 < K) issue_tile(k0 + 32, cur ^ 1);   // prefetch next tile (async DMA)

    v16h af[2], bf[4];
    for (int i = 0; i < 2; ++i) {
      const _Float16* p = As[cur] + (wm + i * 16 + lr) * 32;   // A: e<8 -> K=hf*8+e ; e>=8 -> K=16+hf*8+(e-8)
      v8h lo = *(const v8h*)(p + hf * 8);
      v8h hh = *(const v8h*)(p + 16 + hf * 8);
      af[i] = SHUF16(lo, hh);
    }
    for (int j = 0; j < 4; ++j) {
      const _Float16* p = Bs[cur] + (wn + j * 16 + lr) * 32;   // B: e<8 -> K=hf*8+e ; e>=8 -> K=16+hf*8+(e-8)
      v8h lo = *(const v8h*)(p + hf * 8);
      v8h hh = *(const v8h*)(p + 16 + hf * 8);
      bf[j] = SHUF16(lo, hh);
    }
    for (int i = 0; i < 2; ++i)
      for (int j = 0; j < 4; ++j)
        acc[i][j] = WMMA_F16(af[i], bf[j], acc[i][j]);

    wait_async0();        // own prefetch landed
    __syncthreads();      // everyone's prefetch landed, reads of 'cur' done
  }

  // epilogue (C/D layout: lane = col, vgpr e = row (+8 for upper half-wave))
  for (int i = 0; i < 2; ++i)
    for (int j = 0; j < 4; ++j) {
      int col = n0 + wn + j * 16 + lr;
      for (int e = 0; e < 8; ++e) {
        int row = r0 + wm + i * 16 + e + hf * 8;
        float v = acc[i][j][e];
        size_t o = (size_t)row * N + col;
        if (EPI == 1) {
          ((_Float16*)outv)[o] = (_Float16)v;
        } else if (EPI == 2) {
          float mr = mask[row];
          ((_Float16*)outv)[o] = (_Float16)(gelu_tanh(v) * mr);
        } else if (EPI == 3) {
          ((float*)outv)[o] = res[o] + v;
        } else if (EPI == 4) {
          float mr = mask[row];
          ((float*)outv)[o] = res[o] + v * mr;
        } else {
          ((float*)outv)[o] = v;
        }
      }
    }
}

// ---------------------------------------------------------------- QKV pack
// qkv16: [B,T,3,H,DH] -> q/k: [B,H,T,DH], vT: [B,H,DH,T]
__global__ __launch_bounds__(256) void pack_qkv_kernel(const _Float16* __restrict__ qkv,
                                                       _Float16* __restrict__ q,
                                                       _Float16* __restrict__ k,
                                                       _Float16* __restrict__ vT) {
  int idx = blockIdx.x * 256 + threadIdx.x;       // 0 .. B*T*H*DH-1 (2^22)
  int d = idx & 63;
  int h = (idx >> 6) & 15;
  int t = (idx >> 10) & 2047;
  int b = idx >> 21;
  size_t base = ((size_t)(b * Tt + t)) * (3 * Hh * DH) + h * DH + d;
  _Float16 qv = qkv[base];
  _Float16 kv = qkv[base + Hh * DH];
  _Float16 vv = qkv[base + 2 * Hh * DH];
  size_t qo = (((size_t)(b * Hh + h)) * Tt + t) * DH + d;
  q[qo] = qv;
  k[qo] = kv;
  size_t vo = (((size_t)(b * Hh + h)) * DH + d) * Tt + t;
  vT[vo] = vv;
}

// ---------------------------------------------------------------- Flash attention
// grid: (T/128, H, B); 8 waves per block, each wave owns 16 q rows.
// kv tiles of 32: S = Q K^T via 2 wmma per 16-col block, online softmax,
// P tile bounced through per-wave LDS (D-layout -> A-layout), O += P V via wmma.
__global__ __launch_bounds__(256) void flash_attn_kernel(const _Float16* __restrict__ q16,
                                                         const _Float16* __restrict__ k16,
                                                         const _Float16* __restrict__ vT16,
                                                         const float* __restrict__ mask,
                                                         _Float16* __restrict__ ao16) {
  __shared__ _Float16 plds[8 * 16 * 32];     // 8 KB, per-wave 16x32 P tile
  const int lane = threadIdx.x & 31;
  const int wave = threadIdx.x >> 5;
  const int lr = lane & 15, hf = lane >> 4;
  const int b = blockIdx.z, h = blockIdx.y;
  const int qbase = blockIdx.x * 128 + wave * 16;
  const float NEG_INF = -__builtin_inff();

  const _Float16* qp = q16 + ((size_t)(b * Hh + h)) * Tt * DH;
  const _Float16* kp = k16 + ((size_t)(b * Hh + h)) * Tt * DH;
  const _Float16* vp = vT16 + ((size_t)(b * Hh + h)) * DH * Tt;
  const float* mk = mask + (size_t)b * Tt;

  // Q A-fragments for contraction chunks K = [0,32) and [32,64)
  v16h aq0, aq1;
  {
    const _Float16* r = qp + (size_t)(qbase + lr) * DH;
    v8h lo = *(const v8h*)(r + hf * 8);
    v8h hh = *(const v8h*)(r + 16 + hf * 8);
    aq0 = SHUF16(lo, hh);
    lo = *(const v8h*)(r + 32 + hf * 8);
    hh = *(const v8h*)(r + 48 + hf * 8);
    aq1 = SHUF16(lo, hh);
  }

  v8f o[4];
  for (int d = 0; d < 4; ++d) { v8f z = {}; o[d] = z; }
  float mrow[8], lrow[8];
  for (int i = 0; i < 8; ++i) { mrow[i] = NEG_INF; lrow[i] = 0.f; }
  const float scale = 0.125f;  // DH^-0.5

  for (int kv0 = 0; kv0 < qbase + 16; kv0 += 32) {
    // S = Q K^T, two 16-wide kv blocks
    v8f s[2];
    for (int nb = 0; nb < 2; ++nb) {
      const _Float16* r = kp + (size_t)(kv0 + nb * 16 + lr) * DH;
      v8h lo = *(const v8h*)(r + hf * 8);
      v8h hh = *(const v8h*)(r + 16 + hf * 8);
      v16h bk0 = SHUF16(lo, hh);
      lo = *(const v8h*)(r + 32 + hf * 8);
      hh = *(const v8h*)(r + 48 + hf * 8);
      v16h bk1 = SHUF16(lo, hh);
      v8f z = {};
      z = WMMA_F16(aq0, bk0, z);
      z = WMMA_F16(aq1, bk1, z);
      s[nb] = z;
    }

    // scale + causal + length mask, row max via 16-lane shfl reduction
    int kvc0 = kv0 + lr, kvc1 = kv0 + 16 + lr;
    float mv0 = mk[kvc0], mv1 = mk[kvc1];
    float corr[8];
    for (int i = 0; i < 8; ++i) {
      int qrow = qbase + i + hf * 8;
      float a0 = s[0][i] * scale, a1 = s[1][i] * scale;
      if (kvc0 > qrow || mv0 <= 0.f) a0 = NEG_INF;
      if (kvc1 > qrow || mv1 <= 0.f) a1 = NEG_INF;
      s[0][i] = a0; s[1][i] = a1;
      float r = fmaxf(a0, a1);
      r = fmaxf(r, __shfl_xor(r, 1));
      r = fmaxf(r, __shfl_xor(r, 2));
      r = fmaxf(r, __shfl_xor(r, 4));
      r = fmaxf(r, __shfl_xor(r, 8));
      float mn = fmaxf(mrow[i], r);
      corr[i] = (mn == NEG_INF) ? 1.f : expf(mrow[i] - mn);
      mrow[i] = mn;
    }

    // P = exp(S - m), row sums, write P tile to per-wave LDS (D-layout -> memory)
    _Float16* pb = plds + wave * 512;
    for (int i = 0; i < 8; ++i) {
      int m = i + hf * 8;
      float p0 = (mrow[i] == NEG_INF) ? 0.f : expf(s[0][i] - mrow[i]);
      float p1 = (mrow[i] == NEG_INF) ? 0.f : expf(s[1][i] - mrow[i]);
      float rs = p0 + p1;
      rs += __shfl_xor(rs, 1);
      rs += __shfl_xor(rs, 2);
      rs += __shfl_xor(rs, 4);
      rs += __shfl_xor(rs, 8);
      lrow[i] = lrow[i] * corr[i] + rs;
      pb[m * 32 + lr] = (_Float16)p0;
      pb[m * 32 + 16 + lr] = (_Float16)p1;
    }

    // rescale O accumulator
    for (int i = 0; i < 8; ++i) {
      float c = corr[i];
      o[0][i] *= c; o[1][i] *= c; o[2][i] *= c; o[3][i] *= c;
    }

    // P back as A-fragment (same-wave DS ordering guarantees RAW)
    v8h lo = *(const v8h*)(pb + lr * 32 + hf * 8);
    v8h hh = *(const v8h*)(pb + lr * 32 + 16 + hf * 8);
    v16h pf = SHUF16(lo, hh);

    // O += P @ V   (V^T rows are contiguous along kv -> direct B-fragment loads)
    for (int dhb = 0; dhb < 4; ++dhb) {
      const _Float16* r = vp + (size_t)(dhb * 16 + lr) * Tt + kv0;
      v8h vlo = *(const v8h*)(r + hf * 8);
      v8h vhh = *(const v8h*)(r + 16 + hf * 8);
      v16h bv = SHUF16(vlo, vhh);
      o[dhb] = WMMA_F16(pf, bv, o[dhb]);
    }
  }

  // normalize and store as f16 in [B,T,H*DH]
  for (int dhb = 0; dhb < 4; ++dhb)
    for (int i = 0; i < 8; ++i) {
      int qrow = qbase + i + hf * 8;
      float l = lrow[i];
      float val = (l > 0.f) ? o[dhb][i] / l : 0.f;
      ao16[((size_t)(b * Tt + qrow)) * (Hh * DH) + h * DH + dhb * 16 + lr] = (_Float16)val;
    }
}

// ---------------------------------------------------------------- launch
// ws layout (bytes, 256-aligned regions); total need ~112 MB.
static constexpr size_t OFF_QKVW = 0;                          // 3072*1024 f16
static constexpr size_t OFF_OW   = OFF_QKVW + 6291456;         // 1024*1024 f16
static constexpr size_t OFF_C1W  = OFF_OW   + 2097152;         // 4096*1024 f16
static constexpr size_t OFF_C2W  = OFF_C1W  + 8388608;         // 1024*4096 f16
static constexpr size_t OFF_H16  = OFF_C2W  + 8388608;         // 4096*1024 f16
static constexpr size_t OFF_QKV16 = OFF_H16 + 8388608;         // 4096*3072 f16 (region also reused by ffn16)
static constexpr size_t OFF_FFN16 = OFF_QKV16;                 // 4096*4096 f16 (qkv16 dead by then)
static constexpr size_t OFF_Q16  = OFF_QKV16 + 33554432;       // 2*16*2048*64 f16
static constexpr size_t OFF_K16  = OFF_Q16  + 8388608;
static constexpr size_t OFF_VT16 = OFF_K16  + 8388608;
static constexpr size_t OFF_AO16 = OFF_VT16 + 8388608;
static constexpr size_t OFF_X1   = OFF_AO16 + 8388608;         // 4096*1024 f32

extern "C" void kernel_launch(void* const* d_in, const int* in_sizes, int n_in,
                              void* d_out, int out_size, void* d_ws, size_t ws_size,
                              hipStream_t stream) {
  const float* x       = (const float*)d_in[0];
  const float* mask    = (const float*)d_in[1];
  const float* ln1_w   = (const float*)d_in[2];
  const float* qkv_w   = (const float*)d_in[3];
  const float* o_w     = (const float*)d_in[4];
  const float* ln2_w   = (const float*)d_in[5];
  const float* conv1_w = (const float*)d_in[6];
  const float* conv2_w = (const float*)d_in[7];
  float* out = (float*)d_out;

  char* ws = (char*)d_ws;
  _Float16* qkvw16 = (_Float16*)(ws + OFF_QKVW);
  _Float16* ow16   = (_Float16*)(ws + OFF_OW);
  _Float16* c1w16  = (_Float16*)(ws + OFF_C1W);
  _Float16* c2w16  = (_Float16*)(ws + OFF_C2W);
  _Float16* h16    = (_Float16*)(ws + OFF_H16);
  _Float16* qkv16  = (_Float16*)(ws + OFF_QKV16);
  _Float16* ffn16  = (_Float16*)(ws + OFF_FFN16);
  _Float16* q16    = (_Float16*)(ws + OFF_Q16);
  _Float16* k16    = (_Float16*)(ws + OFF_K16);
  _Float16* vT16   = (_Float16*)(ws + OFF_VT16);
  _Float16* ao16   = (_Float16*)(ws + OFF_AO16);
  float*    x1     = (float*)(ws + OFF_X1);

  // 1) weights -> f16
  f32_to_f16_kernel<<<(3072 * 1024) / 256, 256, 0, stream>>>(qkv_w, qkvw16, 3072 * 1024);
  f32_to_f16_kernel<<<(1024 * 1024) / 256, 256, 0, stream>>>(o_w, ow16, 1024 * 1024);
  f32_to_f16_kernel<<<(4096 * 1024) / 256, 256, 0, stream>>>(conv1_w, c1w16, 4096 * 1024);
  f32_to_f16_kernel<<<(1024 * 4096) / 256, 256, 0, stream>>>(conv2_w, c2w16, 1024 * 4096);

  // 2) h = LN1(x)
  ln_kernel<false><<<Mrows, 256, 0, stream>>>(x, ln1_w, nullptr, h16);

  // 3) qkv = h @ qkv_w^T  (f16 out)
  gemm_wmma<1><<<dim3((3 * Hh * DH) / 128, Mrows / 128), 256, 0, stream>>>(
      h16, qkvw16, qkv16, nullptr, nullptr, Mrows, 3 * Hh * DH, Dd);

  // 4) split/pack q,k,v (v transposed)
  pack_qkv_kernel<<<(Bb * Tt * Hh * DH) / 256, 256, 0, stream>>>(qkv16, q16, k16, vT16);

  // 5) flash attention
  flash_attn_kernel<<<dim3(Tt / 128, Hh, Bb), 256, 0, stream>>>(q16, k16, vT16, mask, ao16);

  // 6) x1 = x + attn @ o_w^T
  gemm_wmma<3><<<dim3(Dd / 128, Mrows / 128), 256, 0, stream>>>(
      ao16, ow16, x1, x, nullptr, Mrows, Dd, Hh * DH);

  // 7) h2 = LN2(x1) * m
  ln_kernel<true><<<Mrows, 256, 0, stream>>>(x1, ln2_w, mask, h16);

  // 8) ffn = gelu(h2 @ conv1_w^T) * m  (f16 out)
  gemm_wmma<2><<<dim3(DFF / 128, Mrows / 128), 256, 0, stream>>>(
      h16, c1w16, ffn16, nullptr, mask, Mrows, DFF, Dd);

  // 9) out = x1 + (ffn @ conv2_w^T) * m
  gemm_wmma<4><<<dim3(Dd / 128, Mrows / 128), 256, 0, stream>>>(
      ffn16, c2w16, out, x1, mask, Mrows, Dd, DFF);
}
